// DeformableTransformerDecoderLayer_24429773980430
// MI455X (gfx1250) — matrix operand order repaired and checked by reading, not compiled, not run on hardware
//
#include <hip/hip_runtime.h>

typedef __bf16 bf16_t;
typedef __attribute__((ext_vector_type(16))) __bf16 v16bf;
typedef __attribute__((ext_vector_type(8)))  float  v8f;

union Frag { v16bf v; uint4 u[2]; };

__device__ __forceinline__ v8f wmma_bf16(v16bf a, v16bf b, v8f c) {
  // emits v_wmma_f32_16x16x32_bf16 (CDNA5 / gfx1250)
  return __builtin_amdgcn_wmma_f32_16x16x32_bf16(false, a, false, b, (short)0, c, false, false);
}

// ---------------------------------------------------------------- elementwise
__global__ void k_f32_to_bf16(const float* __restrict__ x, bf16_t* __restrict__ y, long n) {
  long i = (long)blockIdx.x * blockDim.x + threadIdx.x;
  if (i < n) y[i] = (bf16_t)x[i];
}

__global__ void k_add_bf16(const float* __restrict__ a, const float* __restrict__ b,
                           bf16_t* __restrict__ y, long n) {
  long i = (long)blockIdx.x * blockDim.x + threadIdx.x;
  if (i < n) y[i] = (bf16_t)(a[i] + b[i]);
}

// ---------------------------------------------------------------- generic WMMA GEMM
// out[M,N] = A[M,K](bf16, lda) @ W[N,K]^T(bf16) + bias ; optional relu / f32-residual.
// Each wave computes a 16(M) x 64(N) tile: one A fragment feeds 4 WMMAs per k-step.
// MODE: 0=f32 out, 1=bf16 out, 2=both, 3=bf16 transposed store for V^T (B,H,DH,LQ)
template<int MODE, bool RELU, bool RESID>
__global__ void k_gemm(const bf16_t* __restrict__ A, const bf16_t* __restrict__ W,
                       const float* __restrict__ bias, const float* __restrict__ resid,
                       float* __restrict__ outF, bf16_t* __restrict__ outB,
                       int M, int N, int K, int lda) {
  int wave = (int)(((long)blockIdx.x * blockDim.x + threadIdx.x) >> 5);
  int lane = threadIdx.x & 31;
  int mtiles = M >> 4, ntiles = N >> 6;          // N blocked by 64
  if (wave >= mtiles * ntiles) return;
  int mt = wave / ntiles, nt = wave - mt * ntiles;
  int lm = lane & 15, hi = lane >> 4;
  const bf16_t* Arow = A + (size_t)(mt * 16 + lm) * lda;
  const bf16_t* Wrow = W + (size_t)(nt * 64 + lm) * K;
  const int a0 = hi ? 16 : 0, a1 = hi ? 24 : 8, b0 = hi ? 16 : 0;
  v8f acc[4] = {};
  for (int k = 0; k < K; k += 32) {
    Frag fa;
    fa.u[0] = *(const uint4*)(Arow + k + a0);
    fa.u[1] = *(const uint4*)(Arow + k + a1);
#pragma unroll
    for (int t = 0; t < 4; ++t) {
      Frag fb;
      fb.u[0] = *(const uint4*)(Wrow + (size_t)t * 16 * K + k + b0);
      fb.u[1] = *(const uint4*)(Wrow + (size_t)t * 16 * K + k + b0 + 8);
      acc[t] = wmma_bf16(fa.v, fb.v, acc[t]);
    }
  }
#pragma unroll
  for (int t = 0; t < 4; ++t) {
    int n = nt * 64 + t * 16 + lm;
    float bv = bias ? bias[n] : 0.0f;
#pragma unroll
    for (int j = 0; j < 8; ++j) {
      int m = mt * 16 + j + hi * 8;
      float v = acc[t][j] + bv;
      if (RELU) v = fmaxf(v, 0.0f);
      size_t idx = (size_t)m * N + n;
      if (RESID) v += resid[idx];
      if (MODE == 0 || MODE == 2) outF[idx] = v;
      if (MODE == 1 || MODE == 2) outB[idx] = (bf16_t)v;
      if (MODE == 3) { // V^T: (B=4,H=8,DH=32,LQ=1024), m=b*1024+q, n=h*32+d
        int bb = m >> 10, qq = m & 1023, h = n >> 5, d = n & 31;
        outB[((size_t)((bb * 8 + h) * 32 + d) << 10) + qq] = (bf16_t)v;
      }
    }
  }
}

// ---------------------------------------------------------------- fused attention
// per wave: one (b,h, 16-query tile). scores (16x1024) staged in LDS as bf16.
__global__ void __launch_bounds__(64) k_attn(const bf16_t* __restrict__ q,
                                             const bf16_t* __restrict__ kk,
                                             const bf16_t* __restrict__ vT,
                                             bf16_t* __restrict__ o) {
  __shared__ bf16_t S[2][16][1024]; // 64 KB
  int wib = threadIdx.x >> 5, lane = threadIdx.x & 31;
  int wid = blockIdx.x * 2 + wib;             // 2048 waves total
  int qt = wid & 63, h = (wid >> 6) & 7, b = wid >> 9;
  int lm = lane & 15, hi = lane >> 4;
  const float sc = 0.17677669529663688f;      // 1/sqrt(32)

  // Q fragment (A: 16x32, K-dim = DH = 32)
  const bf16_t* qrow = q + (size_t)(b * 1024 + qt * 16 + lm) * 256 + h * 32;
  Frag fq;
  fq.u[0] = *(const uint4*)(qrow + (hi ? 16 : 0));
  fq.u[1] = *(const uint4*)(qrow + (hi ? 24 : 8));

  // phase 1: scores -> LDS
  for (int kt = 0; kt < 64; ++kt) {
    const bf16_t* krow = kk + (size_t)(b * 1024 + kt * 16 + lm) * 256 + h * 32 + (hi ? 16 : 0);
    Frag fk;
    fk.u[0] = *(const uint4*)(krow);
    fk.u[1] = *(const uint4*)(krow + 8);
    v8f s = {};
    s = wmma_bf16(fq.v, fk.v, s);
#pragma unroll
    for (int j = 0; j < 8; ++j)
      S[wib][j + hi * 8][kt * 16 + lm] = (bf16_t)(s[j] * sc);
  }
  __syncthreads();

  // phase 2: row softmax (lane pair l, l+16 splits a row's 1024 cols)
  bf16_t* row = &S[wib][lm][0];
  int c0 = hi * 512;
  float mx = -1e30f;
  for (int c = c0; c < c0 + 512; ++c) mx = fmaxf(mx, (float)row[c]);
  mx = fmaxf(mx, __shfl_xor(mx, 16, 32));
  float sum = 0.0f;
  for (int c = c0; c < c0 + 512; ++c) {
    float e = __expf((float)row[c] - mx);
    sum += e;
    row[c] = (bf16_t)e;
  }
  sum += __shfl_xor(sum, 16, 32);
  float inv = 1.0f / sum;
  for (int c = c0; c < c0 + 512; ++c) row[c] = (bf16_t)((float)row[c] * inv);
  __syncthreads();

  // phase 3: O = P @ V  (K = 1024 keys, N = 32 channels split in two 16-tiles)
  v8f ac0 = {}, ac1 = {};
  const bf16_t* vb = vT + (size_t)(b * 8 + h) * 32 * 1024;
  const bf16_t* prow = &S[wib][lm][0];
  for (int kt = 0; kt < 32; ++kt) {
    Frag pa;
    pa.u[0] = *(const uint4*)(prow + kt * 32 + (hi ? 16 : 0));
    pa.u[1] = *(const uint4*)(prow + kt * 32 + (hi ? 24 : 8));
    const bf16_t* v0 = vb + (size_t)lm * 1024 + kt * 32 + (hi ? 16 : 0);
    const bf16_t* v1 = vb + (size_t)(16 + lm) * 1024 + kt * 32 + (hi ? 16 : 0);
    Frag fb0, fb1;
    fb0.u[0] = *(const uint4*)v0; fb0.u[1] = *(const uint4*)(v0 + 8);
    fb1.u[0] = *(const uint4*)v1; fb1.u[1] = *(const uint4*)(v1 + 8);
    ac0 = wmma_bf16(pa.v, fb0.v, ac0);
    ac1 = wmma_bf16(pa.v, fb1.v, ac1);
  }
#pragma unroll
  for (int j = 0; j < 8; ++j) {
    size_t m = (size_t)(b * 1024 + qt * 16 + j + hi * 8) * 256 + h * 32;
    o[m + lm]      = (bf16_t)ac0[j];
    o[m + 16 + lm] = (bf16_t)ac1[j];
  }
}

// ---------------------------------------------------------------- LayerNorm (C=256), one wave per row
template<bool OUTF, bool OUTB>
__global__ void k_ln(const float* __restrict__ x, const float* __restrict__ g,
                     const float* __restrict__ be, float* __restrict__ yF,
                     bf16_t* __restrict__ yB, int rows) {
  int wave = (int)(((long)blockIdx.x * blockDim.x + threadIdx.x) >> 5);
  int lane = threadIdx.x & 31;
  if (wave >= rows) return;
  const float* xr = x + (size_t)wave * 256;
  float v[8], s = 0.0f, s2 = 0.0f;
#pragma unroll
  for (int i = 0; i < 8; ++i) { v[i] = xr[lane + i * 32]; s += v[i]; s2 += v[i] * v[i]; }
#pragma unroll
  for (int off = 16; off; off >>= 1) { s += __shfl_xor(s, off, 32); s2 += __shfl_xor(s2, off, 32); }
  float mean = s * (1.0f / 256.0f);
  float var  = s2 * (1.0f / 256.0f) - mean * mean;
  float rs   = rsqrtf(var + 1e-5f);
#pragma unroll
  for (int i = 0; i < 8; ++i) {
    int c = lane + i * 32;
    float y = (v[i] - mean) * rs * g[c] + be[c];
    if (OUTF) yF[(size_t)wave * 256 + c] = y;
    if (OUTB) yB[(size_t)wave * 256 + c] = (bf16_t)y;
  }
}

// ---------------------------------------------------------------- attention-weight softmax (16 per head)
__global__ void k_awsoftmax(float* __restrict__ aw) {
  int t = blockIdx.x * blockDim.x + threadIdx.x; // B*LQ*H rows
  if (t >= 32768) return;
  float* r = aw + (size_t)t * 16;
  float v[16], mx = -1e30f;
#pragma unroll
  for (int i = 0; i < 16; ++i) { v[i] = r[i]; mx = fmaxf(mx, v[i]); }
  float s = 0.0f;
#pragma unroll
  for (int i = 0; i < 16; ++i) { v[i] = __expf(v[i] - mx); s += v[i]; }
  float inv = 1.0f / s;
#pragma unroll
  for (int i = 0; i < 16; ++i) r[i] = v[i] * inv;
}

// ---------------------------------------------------------------- MS-deform bilinear sampling
// one wave per (b,q,h); lane = channel d. value layout (B, LSRC, H, DH) f32.
__global__ void k_deform(const float* __restrict__ ref, const float* __restrict__ off,
                         const float* __restrict__ aw, const float* __restrict__ value,
                         bf16_t* __restrict__ out) {
  int wave = (int)(((long)blockIdx.x * blockDim.x + threadIdx.x) >> 5);
  int d = threadIdx.x & 31;
  if (wave >= 32768) return;
  int h = wave & 7, bq = wave >> 3, b = bq >> 10;
  const float* refp = ref + (size_t)bq * 8;            // (L,2)
  const float* offp = off + (size_t)bq * 256 + h * 32; // (L,P,2) for this head
  const float* awp  = aw + (size_t)wave * 16;
  const int Wl[4] = {128, 64, 32, 16};
  const int st[4] = {0, 16384, 20480, 21504};
  float acc = 0.0f;
#pragma unroll
  for (int l = 0; l < 4; ++l) {
    float rx = refp[l * 2 + 0], ry = refp[l * 2 + 1];
    int w = Wl[l];
    float fw = (float)w;
    const float* vl = value + ((size_t)(b * 21760 + st[l])) * 256 + h * 32 + d;
#pragma unroll
    for (int p = 0; p < 4; ++p) {
      float ox = offp[l * 8 + p * 2 + 0], oy = offp[l * 8 + p * 2 + 1];
      float x = (rx + ox / fw) * fw - 0.5f;
      float y = (ry + oy / fw) * fw - 0.5f;
      float fx = floorf(x), fy = floorf(y);
      int x0 = (int)fx, y0 = (int)fy;
      float wx1 = x - fx, wx0 = 1.0f - wx1;
      float wy1 = y - fy, wy0 = 1.0f - wy1;
      float a = awp[l * 4 + p];
      if (x0 >= 0     && x0 < w     && y0 >= 0     && y0 < w)
        acc += a * wx0 * wy0 * vl[(size_t)(y0 * w + x0) * 256];
      if (x0 + 1 >= 0 && x0 + 1 < w && y0 >= 0     && y0 < w)
        acc += a * wx1 * wy0 * vl[(size_t)(y0 * w + x0 + 1) * 256];
      if (x0 >= 0     && x0 < w     && y0 + 1 >= 0 && y0 + 1 < w)
        acc += a * wx0 * wy1 * vl[(size_t)((y0 + 1) * w + x0) * 256];
      if (x0 + 1 >= 0 && x0 + 1 < w && y0 + 1 >= 0 && y0 + 1 < w)
        acc += a * wx1 * wy1 * vl[(size_t)((y0 + 1) * w + x0 + 1) * 256];
    }
  }
  out[(size_t)bq * 256 + h * 32 + d] = (bf16_t)acc;
}

// ---------------------------------------------------------------- host launcher
template<int MODE, bool RELU, bool RESID>
static void gemm(const bf16_t* A, const bf16_t* W, const float* bias, const float* resid,
                 float* oF, bf16_t* oB, int M, int N, int K, int lda, hipStream_t s) {
  long waves = (long)(M / 16) * (N / 64);
  int blocks = (int)((waves * 32 + 255) / 256);
  k_gemm<MODE, RELU, RESID><<<blocks, 256, 0, s>>>(A, W, bias, resid, oF, oB, M, N, K, lda);
}

static void cvt(const float* x, bf16_t* y, long n, hipStream_t s) {
  k_f32_to_bf16<<<(int)((n + 255) / 256), 256, 0, s>>>(x, y, n);
}

extern "C" void kernel_launch(void* const* d_in, const int* in_sizes, int n_in,
                              void* d_out, int out_size, void* d_ws, size_t ws_size,
                              hipStream_t stream) {
  const long M1 = 4096;        // B*LQ
  const long MS = 87040;       // B*LSRC
  const int  C  = 256, DFF = 1024;

  const float* tgt   = (const float*)d_in[0];
  const float* qpos  = (const float*)d_in[1];
  const float* refp  = (const float*)d_in[2];
  const float* src   = (const float*)d_in[3];
  const float* in_w  = (const float*)d_in[4];
  const float* in_b  = (const float*)d_in[5];
  const float* sa_w  = (const float*)d_in[6];
  const float* sa_b  = (const float*)d_in[7];
  const float* off_w = (const float*)d_in[8];
  const float* off_b = (const float*)d_in[9];
  const float* aw_w  = (const float*)d_in[10];
  const float* aw_b  = (const float*)d_in[11];
  const float* val_w = (const float*)d_in[12];
  const float* val_b = (const float*)d_in[13];
  const float* co_w  = (const float*)d_in[14];
  const float* co_b  = (const float*)d_in[15];
  const float* ln1_g = (const float*)d_in[16];
  const float* ln1_b = (const float*)d_in[17];
  const float* ln2_g = (const float*)d_in[18];
  const float* ln2_b = (const float*)d_in[19];
  const float* ln3_g = (const float*)d_in[20];
  const float* ln3_b = (const float*)d_in[21];
  const float* f1_w  = (const float*)d_in[22];
  const float* f1_b  = (const float*)d_in[23];
  const float* f2_w  = (const float*)d_in[24];
  const float* f2_b  = (const float*)d_in[25];
  float* out = (float*)d_out;

  // deterministic workspace carve
  char* p = (char*)d_ws;
  auto take = [&](size_t bytes) -> char* {
    char* r = p; p += (bytes + 255) & ~(size_t)255; return r;
  };
  bf16_t* in_w_bf  = (bf16_t*)take(768 * 256 * 2);
  bf16_t* sa_w_bf  = (bf16_t*)take(256 * 256 * 2);
  bf16_t* off_w_bf = (bf16_t*)take(256 * 256 * 2);
  bf16_t* aw_w_bf  = (bf16_t*)take(128 * 256 * 2);
  bf16_t* val_w_bf = (bf16_t*)take(256 * 256 * 2);
  bf16_t* co_w_bf  = (bf16_t*)take(256 * 256 * 2);
  bf16_t* f1_w_bf  = (bf16_t*)take((size_t)DFF * 256 * 2);
  bf16_t* f2_w_bf  = (bf16_t*)take((size_t)256 * DFF * 2);
  bf16_t* qk_bf    = (bf16_t*)take(M1 * C * 2);
  bf16_t* tgt_bf   = (bf16_t*)take(M1 * C * 2);
  bf16_t* q_bf     = (bf16_t*)take(M1 * C * 2);
  bf16_t* k_bf     = (bf16_t*)take(M1 * C * 2);
  bf16_t* vT_bf    = (bf16_t*)take(M1 * C * 2);
  bf16_t* attno_bf = (bf16_t*)take(M1 * C * 2);
  bf16_t* q2_bf    = (bf16_t*)take(M1 * C * 2);
  bf16_t* samp_bf  = (bf16_t*)take(M1 * C * 2);
  bf16_t* tgt2_bf  = (bf16_t*)take(M1 * C * 2);
  bf16_t* src_bf   = (bf16_t*)take(MS * C * 2);
  bf16_t* ffn1_bf  = (bf16_t*)take(M1 * DFF * 2);
  float*  mha_pre  = (float*)take(M1 * C * 4);
  float*  tgt1_f   = (float*)take(M1 * C * 4);
  float*  offs_f   = (float*)take(M1 * 256 * 4);
  float*  aw_f     = (float*)take(M1 * 128 * 4);
  float*  value_f  = (float*)take(MS * C * 4);
  float*  def_pre  = (float*)take(M1 * C * 4);
  float*  tgt2_f   = (float*)take(M1 * C * 4);
  float*  ffn_pre  = (float*)take(M1 * C * 4);
  (void)ws_size; (void)in_sizes; (void)n_in; (void)out_size;

  // weight conversions
  cvt(in_w,  in_w_bf,  768 * 256, stream);
  cvt(sa_w,  sa_w_bf,  256 * 256, stream);
  cvt(off_w, off_w_bf, 256 * 256, stream);
  cvt(aw_w,  aw_w_bf,  128 * 256, stream);
  cvt(val_w, val_w_bf, 256 * 256, stream);
  cvt(co_w,  co_w_bf,  256 * 256, stream);
  cvt(f1_w,  f1_w_bf,  (long)DFF * 256, stream);
  cvt(f2_w,  f2_w_bf,  (long)256 * DFF, stream);

  // ---- self-attention ----
  k_add_bf16<<<(int)((M1 * C + 255) / 256), 256, 0, stream>>>(tgt, qpos, qk_bf, M1 * C);
  cvt(tgt, tgt_bf, M1 * C, stream);
  gemm<1, false, false>(qk_bf,  in_w_bf,             in_b,       nullptr, nullptr, q_bf,  (int)M1, C, C, C, stream);
  gemm<1, false, false>(qk_bf,  in_w_bf + 256 * 256, in_b + 256, nullptr, nullptr, k_bf,  (int)M1, C, C, C, stream);
  gemm<3, false, false>(tgt_bf, in_w_bf + 512 * 256, in_b + 512, nullptr, nullptr, vT_bf, (int)M1, C, C, C, stream);
  k_attn<<<1024, 64, 0, stream>>>(q_bf, k_bf, vT_bf, attno_bf);
  gemm<0, false, true>(attno_bf, sa_w_bf, sa_b, tgt, mha_pre, nullptr, (int)M1, C, C, C, stream);
  k_ln<true, false><<<(int)((M1 * 32 + 255) / 256), 256, 0, stream>>>(mha_pre, ln2_g, ln2_b, tgt1_f, nullptr, (int)M1);

  // ---- multi-scale deformable attention ----
  k_add_bf16<<<(int)((M1 * C + 255) / 256), 256, 0, stream>>>(tgt1_f, qpos, q2_bf, M1 * C);
  gemm<0, false, false>(q2_bf, off_w_bf, off_b, nullptr, offs_f, nullptr, (int)M1, 256, C, C, stream);
  gemm<0, false, false>(q2_bf, aw_w_bf,  aw_b,  nullptr, aw_f,   nullptr, (int)M1, 128, C, C, stream);
  k_awsoftmax<<<128, 256, 0, stream>>>(aw_f);
  cvt(src, src_bf, MS * C, stream);
  gemm<0, false, false>(src_bf, val_w_bf, val_b, nullptr, value_f, nullptr, (int)MS, C, C, C, stream);
  k_deform<<<4096, 256, 0, stream>>>(refp, offs_f, aw_f, value_f, samp_bf);
  gemm<0, false, true>(samp_bf, co_w_bf, co_b, tgt1_f, def_pre, nullptr, (int)M1, C, C, C, stream);
  k_ln<true, true><<<(int)((M1 * 32 + 255) / 256), 256, 0, stream>>>(def_pre, ln1_g, ln1_b, tgt2_f, tgt2_bf, (int)M1);

  // ---- FFN ----
  gemm<1, true, false>(tgt2_bf, f1_w_bf, f1_b, nullptr, nullptr, ffn1_bf, (int)M1, DFF, C, C, stream);
  gemm<0, false, true>(ffn1_bf, f2_w_bf, f2_b, tgt2_f, ffn_pre, nullptr, (int)M1, C, DFF, DFF, stream);
  k_ln<true, false><<<(int)((M1 * 32 + 255) / 256), 256, 0, stream>>>(ffn_pre, ln3_g, ln3_b, out, nullptr, (int)M1);
}
